// WindowAttention3D_50543175139542
// MI455X (gfx1250) — matrix operand order repaired and checked by reading, not compiled, not run on hardware
//
#include <hip/hip_runtime.h>

// ---------------------------------------------------------------------------
// Swin window attention, fused flash-style, CDNA5 WMMA f16 path + TDM staging.
//   x:(512,343,128) f32, mask:(64,343,343) f32, Wqkv:(128,384), bqkv:(384),
//   Wproj:(128,128), bproj:(128)  ->  out:(512,343,128) f32
// ---------------------------------------------------------------------------

typedef __attribute__((ext_vector_type(16))) _Float16 v16h;
typedef __attribute__((ext_vector_type(8)))  float    v8f;

typedef unsigned int u32x4 __attribute__((ext_vector_type(4)));
typedef int          i32x4 __attribute__((ext_vector_type(4)));
typedef int          i32x8 __attribute__((ext_vector_type(8)));

union Frag16 {
  v16h h;
  unsigned int u[8];
  _Float16 e[16];
};

#define SCALE_Q 0.17677669529663687f   // 32^-0.5

__device__ __forceinline__ v8f wmma_f16(v16h a, v16h b, v8f c) {
  return __builtin_amdgcn_wmma_f32_16x16x32_f16(false, a, false, b, (short)0, c,
                                                false, false);
}

// A-fragment (16x32 f16, M x K), src row-major with row stride `ld` halfs.
// Lane layout: m = lane&15; VGPR j holds K pair
//   k0 = ((lane>>4)<<4) + (j<<1)
__device__ __forceinline__ v16h load_a_frag(const _Float16* src, int ld) {
  int l = threadIdx.x & 31;
  int m = l & 15;
  int kh = (l >> 4) << 4;
  Frag16 f;
  const unsigned short* s = (const unsigned short*)src;
#pragma unroll
  for (int j = 0; j < 8; ++j) {
    int k0 = kh + (j << 1);
    f.u[j] = *(const unsigned int*)(s + m * ld + k0);
  }
  return f.h;
}

// B-fragment (32x16 f16, K x N) from a row-major f32 weight block W[k][n],
// row stride `ld` floats. Lane layout: n = lane&15; K = ((lane>>4)<<4)+2j+{0,1}.
__device__ __forceinline__ v16h load_b_frag_w32(const float* src, int ld) {
  int l = threadIdx.x & 31;
  int n = l & 15;
  int kh = (l >> 4) << 4;
  Frag16 f;
#pragma unroll
  for (int j = 0; j < 8; ++j) {
    int k = kh + 2 * j;
    f.e[2 * j]     = (_Float16)src[k * ld + n];
    f.e[2 * j + 1] = (_Float16)src[(k + 1) * ld + n];
  }
  return f.h;
}

__device__ __forceinline__ unsigned int pack_h2(float a, float b) {
  union { _Float16 h[2]; unsigned int u; } t;
  t.h[0] = (_Float16)a;
  t.h[1] = (_Float16)b;
  return t.u;
}

// ---------------------------------------------------------------------------
// Tensor Data Mover: 2D tile load (f16 elements) global -> LDS.
//   lds_addr     : LDS byte address of tile start
//   gaddr        : global address of tile start
//   tensor_d0/d1 : tensor extents (elements) for OOB zero-fill
//   stride0      : elements between consecutive dim-1 lines
//   tile_d0/d1   : tile extents (elements)
//   pad_*        : optional LDS row padding (interval/amount are ISA codes)
// ---------------------------------------------------------------------------
#if defined(__has_builtin)
#if __has_builtin(__builtin_amdgcn_tensor_load_to_lds)
#define HAVE_TDM 1
#endif
#endif

__device__ __forceinline__ void tdm_load_2d_f16(
    unsigned lds_addr, const void* gaddr, unsigned tensor_d0,
    unsigned tensor_d1, unsigned stride0, unsigned tile_d0, unsigned tile_d1,
    unsigned pad_en, unsigned pad_interval, unsigned pad_amount) {
#ifdef HAVE_TDM
  unsigned long long ga = (unsigned long long)gaddr;
  union { u32x4 v; unsigned a[4]; } g0;
  g0.a[0] = 1u;                                       // count=1, user mode
  g0.a[1] = lds_addr;                                 // bits 63:32
  g0.a[2] = (unsigned)(ga & 0xffffffffu);             // global_addr lo
  g0.a[3] = (unsigned)((ga >> 32) & 0x01ffffffu) |    // global_addr hi (57b)
            (2u << 30);                               // type=2 ("image")
  union { i32x8 v; unsigned a[8]; } g1;
  g1.a[0] = (1u << 16) |                              // data_size = 2 bytes
            (pad_en << 20) | (pad_interval << 22) | (pad_amount << 25);
  g1.a[1] = tensor_d0 << 16;                          // tensor_dim0 lo16
  g1.a[2] = (tensor_d0 >> 16) | (tensor_d1 << 16);    // dim0 hi | dim1 lo
  g1.a[3] = (tensor_d1 >> 16) | (tile_d0 << 16);      // dim1 hi | tile_dim0
  g1.a[4] = tile_d1 & 0xffffu;                        // tile_dim1 (dim2=0)
  g1.a[5] = stride0;                                  // tensor_dim0_stride lo
  g1.a[6] = 0u;                                       // stride0 hi | stride1 lo
  g1.a[7] = 0u;
  i32x4 z4 = {};
#if __clang_major__ >= 23
  i32x8 z8 = {};
  __builtin_amdgcn_tensor_load_to_lds(g0.v, g1.v, z4, z4, z8, 0);
#else
  __builtin_amdgcn_tensor_load_to_lds(g0.v, g1.v, z4, z4, 0);
#endif
#endif
}

// ---------------------------------------------------------------------------
// Kernel 1: qkv = x @ Wqkv + bqkv  (fp32 in, f16 out, rearranged)
//   Q_ws[bw][h][n][32]   (pre-scaled by SCALE_Q)
//   K_ws[bw][h][n][32]
//   VT_ws[bw][h][32][344] (transposed, padded row stride)
// grid 10976 blocks x 384 thr (12 waves); block = 16 tokens, wave = 2 col tiles
// ---------------------------------------------------------------------------
__global__ __launch_bounds__(384) void qkv_kernel(
    const float* __restrict__ x, const float* __restrict__ Wqkv,
    const float* __restrict__ bqkv, _Float16* __restrict__ Qw,
    _Float16* __restrict__ Kw, _Float16* __restrict__ VTw) {
  __shared__ __align__(16) _Float16 xs[16 * 136];
  const int tid = threadIdx.x;
  const int t0 = blockIdx.x * 16;

  for (int idx = tid; idx < 2048; idx += 384) {
    int r = idx >> 7, c = idx & 127;
    xs[r * 136 + c] = (_Float16)x[(t0 + r) * 128 + c];
  }
  __syncthreads();

  const int wv = tid >> 5;
  const int l = tid & 31;

  v16h a[4];
#pragma unroll
  for (int kc = 0; kc < 4; ++kc) a[kc] = load_a_frag(xs + kc * 32, 136);

#pragma unroll
  for (int cti = 0; cti < 2; ++cti) {
    const int ct = wv * 2 + cti;            // 0..23 col tiles of 16 over N=384
    v8f acc = {};
#pragma unroll
    for (int kc = 0; kc < 4; ++kc) {
      v16h b = load_b_frag_w32(Wqkv + kc * 32 * 384 + ct * 16, 384);
      acc = wmma_f16(a[kc], b, acc);
    }
    const int col = ct * 16 + (l & 15);
    const float bias = bqkv[col];
    const int sec = ct >> 3;                // 0=Q 1=K 2=V
    const int h = (ct >> 1) & 3;
    const int d = ((ct & 1) << 4) + (l & 15);
#pragma unroll
    for (int j = 0; j < 8; ++j) {
      const int m = ((l >> 4) << 3) + j;
      const int t = t0 + m;                 // 0..175615, always valid
      const int bw = t / 343;
      const int n = t - bw * 343;
      const float v = acc[j] + bias;
      if (sec == 0) {
        Qw[((bw * 4 + h) * 343 + n) * 32 + d] = (_Float16)(v * SCALE_Q);
      } else if (sec == 1) {
        Kw[((bw * 4 + h) * 343 + n) * 32 + d] = (_Float16)v;
      } else {
        VTw[((bw * 4 + h) * 32 + d) * 344 + n] = (_Float16)v;
      }
    }
  }
}

// ---------------------------------------------------------------------------
// Kernel 2: fused flash attention per (window, head).
//   S^T = K(16x32) x Q^T(32x16)  ->  softmax stats per-lane (q = lane column)
//   O^T += V^T(16x32) x P^T(32x16)
// K/V^T staged to LDS by the Tensor Data Mover (TDM zero-fills OOB tails and
// inserts the bank-conflict padding for K's rows).
// grid 512*4*2 blocks x 352 thr (11 waves); wave = one 16-row q tile.
// ---------------------------------------------------------------------------
__global__ __launch_bounds__(352) void attn_kernel(
    const float* __restrict__ mask, const _Float16* __restrict__ Qw,
    const _Float16* __restrict__ Kw, const _Float16* __restrict__ VTw,
    _Float16* __restrict__ Ow) {
  __shared__ __align__(16) _Float16 Kls[352 * 36];    // [tok][d], pad stride 36
  __shared__ __align__(16) _Float16 VTls[32 * 352];   // [d][tok], stride 352

  const int bx = blockIdx.x;
  const int part = bx & 1;
  const int h = (bx >> 1) & 3;
  const int bw = bx >> 3;
  const int tid = threadIdx.x;

  const _Float16* Qh = Qw + (size_t)(bw * 4 + h) * 343 * 32;
  const _Float16* Kh = Kw + (size_t)(bw * 4 + h) * 343 * 32;
  const _Float16* VTh = VTw + (size_t)(bw * 4 + h) * 32 * 344;

  const int wv = tid >> 5;
  const int l = tid & 31;

#ifdef HAVE_TDM
  // LDS aperture flat addresses carry the LDS byte offset in addr[31:0].
  if (wv == 0) {
    // K: 343x32 tensor, 352x32 tile (rows 343..351 zero-filled by OOB),
    // LDS rows padded 16DW + 2DW -> stride 36 halfs (codes: interval=3, amt=1)
    tdm_load_2d_f16((unsigned)(unsigned long long)(void*)&Kls[0], Kh,
                    /*tensor_d0=*/32, /*tensor_d1=*/343, /*stride0=*/32,
                    /*tile_d0=*/32, /*tile_d1=*/352,
                    /*pad_en=*/1, /*pad_interval=*/3, /*pad_amount=*/1);
  } else if (wv == 1) {
    // V^T: 343x32 tensor (dim0 = tokens), 352x32 tile -> cols 343..351 zeroed.
    tdm_load_2d_f16((unsigned)(unsigned long long)(void*)&VTls[0], VTh,
                    /*tensor_d0=*/343, /*tensor_d1=*/32, /*stride0=*/344,
                    /*tile_d0=*/352, /*tile_d1=*/32,
                    /*pad_en=*/0, /*pad_interval=*/0, /*pad_amount=*/0);
  }
  __builtin_amdgcn_s_wait_tensorcnt(0);
#else
  {  // fallback: manual staging
    const unsigned int* src = (const unsigned int*)Kh;
    unsigned int* dst = (unsigned int*)Kls;
    for (int idx = tid; idx < 352 * 16; idx += 352) {
      int r = idx >> 4, c = idx & 15;
      dst[r * 18 + c] = (r < 343) ? src[r * 16 + c] : 0u;
    }
    const unsigned int* src2 = (const unsigned int*)VTh;
    unsigned int* dst2 = (unsigned int*)VTls;
    for (int idx = tid; idx < 32 * 176; idx += 352) {
      int r = idx / 176, c = idx - r * 176;
      dst2[r * 176 + c] = (c < 172) ? src2[r * 172 + c] : 0u;
    }
  }
#endif
  __syncthreads();

  const int qbase = (part * 11 + wv) * 16;  // q tile: 0..21
  const int lane16 = l & 15;
  const int hf = l >> 4;
  const int qcol = qbase + lane16;          // this lane's q column
  const int qc = qcol < 343 ? qcol : 342;   // clamped for loads

  // Q^T B-fragment (K=d contiguous pairs in Q_ws row) — loaded once.
  v16h bq;
  {
    Frag16 f;
    const unsigned short* s =
        (const unsigned short*)(Qh + qc * 32 + (hf << 4));
#pragma unroll
    for (int j = 0; j < 8; ++j) f.u[j] = *(const unsigned int*)(s + 2 * j);
    bq = f.h;
  }

  const float* mq = mask + (size_t)(bw & 63) * 343 * 343 + (size_t)qc * 343;

  float m_run = -3.0e38f, l_run = 0.f;
  v8f acc0 = {}, acc1 = {};
  const v8f zero = {};

  for (int kb = 0; kb < 352; kb += 32) {
    v16h ak0 = load_a_frag(Kls + kb * 36, 36);
    v16h ak1 = load_a_frag(Kls + (kb + 16) * 36, 36);
    v8f s0 = wmma_f16(ak0, bq, zero);   // S^T rows kb..kb+15
    v8f s1 = wmma_f16(ak1, bq, zero);   // S^T rows kb+16..kb+31

    float p0[8], p1[8];
#pragma unroll
    for (int j = 0; j < 8; ++j) {
      const int k0 = kb + hf * 8 + j;
      const int k1 = k0 + 16;
      p0[j] = (k0 < 343) ? s0[j] + mq[k0] : -1e30f;
      p1[j] = (k1 < 343) ? s1[j] + mq[k1] : -1e30f;
    }
    float mx = -3.0e38f;
#pragma unroll
    for (int j = 0; j < 8; ++j) mx = fmaxf(mx, fmaxf(p0[j], p1[j]));
    mx = fmaxf(mx, __shfl_xor(mx, 16));

    const float m_new = fmaxf(m_run, mx);
    const float corr = __expf(m_run - m_new);
    float rs = 0.f;
#pragma unroll
    for (int j = 0; j < 8; ++j) {
      p0[j] = __expf(p0[j] - m_new);
      p1[j] = __expf(p1[j] - m_new);
      rs += p0[j] + p1[j];
    }
    rs += __shfl_xor(rs, 16);
    l_run = l_run * corr + rs;
    m_run = m_new;
#pragma unroll
    for (int j = 0; j < 8; ++j) {
      acc0[j] = acc0[j] * corr;
      acc1[j] = acc1[j] * corr;
    }

    // C-layout P^T -> B-layout via one half exchange.
    float lo[8], hi[8];
#pragma unroll
    for (int j = 0; j < 8; ++j) {
      const float c = hf ? p0[j] : p1[j];
      const float dx = __shfl_xor(c, 16);
      lo[j] = hf ? dx : p0[j];
      hi[j] = hf ? p1[j] : dx;
    }
    Frag16 bp;
#pragma unroll
    for (int j = 0; j < 4; ++j) bp.u[j] = pack_h2(lo[2 * j], lo[2 * j + 1]);
#pragma unroll
    for (int j = 4; j < 8; ++j)
      bp.u[j] = pack_h2(hi[2 * (j - 4)], hi[2 * (j - 4) + 1]);

    v16h av0 = load_a_frag(VTls + kb, 352);             // d = 0..15
    v16h av1 = load_a_frag(VTls + 16 * 352 + kb, 352);  // d = 16..31
    acc0 = wmma_f16(av0, bp.h, acc0);
    acc1 = wmma_f16(av1, bp.h, acc1);
  }

  const float inv = 1.0f / l_run;
  if (qcol < 343) {
    _Float16* dst = Ow + ((size_t)bw * 343 + qcol) * 128 + h * 32;
#pragma unroll
    for (int j = 0; j < 8; j += 2) {
      const int d = hf * 8 + j;
      *(unsigned int*)(dst + d) = pack_h2(acc0[j] * inv, acc0[j + 1] * inv);
      *(unsigned int*)(dst + 16 + d) =
          pack_h2(acc1[j] * inv, acc1[j + 1] * inv);
    }
  }
}

// ---------------------------------------------------------------------------
// Kernel 3: out = O @ Wproj + bproj  (f16 in, f32 out)
// grid 10976 blocks x 256 thr (8 waves); block = 16 tokens, wave = 1 col tile
// ---------------------------------------------------------------------------
__global__ __launch_bounds__(256) void proj_kernel(
    const _Float16* __restrict__ Ow, const float* __restrict__ Wp,
    const float* __restrict__ bp, float* __restrict__ out) {
  __shared__ __align__(16) _Float16 xs[16 * 136];
  const int tid = threadIdx.x;
  const int t0 = blockIdx.x * 16;

  {
    const unsigned int* src = (const unsigned int*)(Ow + (size_t)t0 * 128);
    unsigned int* dst = (unsigned int*)xs;
    for (int idx = tid; idx < 16 * 64; idx += 256) {
      int r = idx >> 6, c = idx & 63;
      dst[r * 68 + c] = src[r * 64 + c];
    }
  }
  __syncthreads();

  const int wv = tid >> 5;
  const int l = tid & 31;

  v16h a[4];
#pragma unroll
  for (int kc = 0; kc < 4; ++kc) a[kc] = load_a_frag(xs + kc * 32, 136);

  v8f acc = {};
#pragma unroll
  for (int kc = 0; kc < 4; ++kc) {
    v16h b = load_b_frag_w32(Wp + kc * 32 * 128 + wv * 16, 128);
    acc = wmma_f16(a[kc], b, acc);
  }
  const int col = wv * 16 + (l & 15);
  const float bias = bp[col];
#pragma unroll
  for (int j = 0; j < 8; ++j) {
    const int m = ((l >> 4) << 3) + j;
    out[(size_t)(t0 + m) * 128 + col] = acc[j] + bias;
  }
}

// ---------------------------------------------------------------------------
extern "C" void kernel_launch(void* const* d_in, const int* in_sizes, int n_in,
                              void* d_out, int out_size, void* d_ws,
                              size_t ws_size, hipStream_t stream) {
  const float* x = (const float*)d_in[0];
  const float* mask = (const float*)d_in[1];
  const float* Wqkv = (const float*)d_in[2];
  const float* bqkv = (const float*)d_in[3];
  const float* Wproj = (const float*)d_in[4];
  const float* bproj = (const float*)d_in[5];
  float* out = (float*)d_out;

  char* ws = (char*)d_ws;
  const size_t qk_bytes = (size_t)512 * 4 * 343 * 32 * 2;   // ~45 MB each
  const size_t vt_bytes = (size_t)512 * 4 * 32 * 344 * 2;
  _Float16* Qw = (_Float16*)ws;
  _Float16* Kw = (_Float16*)(ws + qk_bytes);
  _Float16* VTw = (_Float16*)(ws + 2 * qk_bytes);
  _Float16* Ow = (_Float16*)(ws + 2 * qk_bytes + vt_bytes);

  qkv_kernel<<<10976, 384, 0, stream>>>(x, Wqkv, bqkv, Qw, Kw, VTw);
  attn_kernel<<<512 * 4 * 2, 352, 0, stream>>>(mask, Qw, Kw, VTw, Ow);
  proj_kernel<<<10976, 256, 0, stream>>>(Ow, Wproj, bproj, out);
}
